// ParallelMambaVIX_59820304499176
// MI455X (gfx1250) — matrix operand order repaired and checked by reading, not compiled, not run on hardware
//
#include <hip/hip_runtime.h>

// ---------------------------------------------------------------------------
// MI455X (gfx1250): wave32, WMMA f32_16x16x32_f16 GEMMs fed from LDS tiles
// staged with GLOBAL_LOAD_ASYNC_TO_LDS (ASYNCcnt) when available; f16
// activations/weights produced once by the elementwise producers.
// ---------------------------------------------------------------------------

typedef __attribute__((ext_vector_type(16))) _Float16 v16h;
typedef __attribute__((ext_vector_type(8)))  _Float16 v8h;
typedef __attribute__((ext_vector_type(4)))  _Float16 v4h;
typedef __attribute__((ext_vector_type(8)))  float    v8f;
typedef __attribute__((ext_vector_type(2)))  int      v2i;

#if defined(__has_builtin)
#  if __has_builtin(__builtin_amdgcn_global_load_async_to_lds_b64)
#    define HAVE_ASYNC_LDS 1
#  endif
#endif
#ifndef HAVE_ASYNC_LDS
#  define HAVE_ASYNC_LDS 0
#endif

__device__ __forceinline__ void copy8_to_lds(const _Float16* __restrict__ gsrc,
                                             _Float16* __restrict__ ldst) {
#if HAVE_ASYNC_LDS
    // 8-byte async copy global->LDS (VGLOBAL async encoding, ASYNCcnt).
    // Param type per hipcc diagnostic: generic pointer to int __vector(2).
    __builtin_amdgcn_global_load_async_to_lds_b64((v2i*)gsrc, (v2i*)ldst, 0, 0);
#else
    *(v4h*)ldst = *(const v4h*)gsrc;
#endif
}

__device__ __forceinline__ void tile_join() {
#if HAVE_ASYNC_LDS
#  if defined(__has_builtin) && __has_builtin(__builtin_amdgcn_s_wait_asynccnt)
    __builtin_amdgcn_s_wait_asynccnt(0);
#  else
    asm volatile("s_wait_asynccnt 0x0" ::: "memory");
#  endif
#endif
    __syncthreads();
}

__device__ __forceinline__ float gelu_exact(float x) {
    return 0.5f * x * (1.0f + erff(x * 0.70710678118654752f));
}
__device__ __forceinline__ float siluf(float x) {
    return x / (1.0f + __expf(-x));
}
__device__ __forceinline__ float softplusf(float x) {
    return (x > 20.0f) ? x : log1pf(__expf(x));
}

__device__ __forceinline__ float blk_sum256(float v, float* red) {
    int tid = threadIdx.x;
    red[tid] = v;
    __syncthreads();
    for (int o = 128; o > 0; o >>= 1) {
        if (tid < o) red[tid] += red[tid + o];
        __syncthreads();
    }
    float r = red[0];
    __syncthreads();
    return r;
}

// ---------------------------------------------------------------------------
// WMMA GEMM: out[M,N] = A[M,K] @ W[N,K]^T (+bias). A,W are f16; out f32.
// block = 256 thr = 8 waves (2 M-tiles x 4 N-tiles); per K-chunk of 32 the
// block stages A-tile (32x32 f16, 2KB) and B-tile (64x32 f16, 4KB) into LDS
// via async copies, then every wave feeds its fragments from LDS.
// Fragment layouts used here (f16, 16x16x32; same contiguous k order for A and B):
//   A: lane<16 -> row=lane, elems 0..15 = K+0..15;
//      lane>=16 -> row=lane-16, elems 0..15 = K+16..31.
//   B: lane&15 = column; lanes 0..15 hold K+0..15, lanes 16..31 K+16..31.
// ---------------------------------------------------------------------------
template <int K>
__global__ void k_gemm_wmma(const _Float16* __restrict__ A,
                            const _Float16* __restrict__ W,
                            const float* __restrict__ bias,
                            float* __restrict__ out, int M, int N) {
    __shared__ _Float16 sA[32 * 32];
    __shared__ _Float16 sB[64 * 32];
    const int tid  = threadIdx.x;
    const int wave = tid >> 5;
    const int lane = tid & 31;
    const int wy = wave >> 2, wx = wave & 3;
    const int tMblk = blockIdx.x * 32;
    const int tNblk = blockIdx.y * 64;
    const int l15 = lane & 15;
    const int hi  = lane >> 4;

    // copy lanes: thread -> (row 0..31, 4-half chunk 0..7)
    const int cr = tid >> 3;
    const int cc = (tid & 7) * 4;
    const _Float16* aSrc = A + (size_t)(tMblk + cr) * K + cc;
    int col0 = tNblk + cr;      if (col0 >= N) col0 = N - 1;
    int col1 = tNblk + 32 + cr; if (col1 >= N) col1 = N - 1;
    const _Float16* bSrc0 = W + (size_t)col0 * K + cc;
    const _Float16* bSrc1 = W + (size_t)col1 * K + cc;
    _Float16* aDst  = &sA[cr * 32 + cc];
    _Float16* bDst0 = &sB[cr * 32 + cc];
    _Float16* bDst1 = &sB[(cr + 32) * 32 + cc];

    const _Float16* aFrag = &sA[(wy * 16 + l15) * 32 + hi * 16];
    const _Float16* bFrag = &sB[(wx * 16 + l15) * 32 + hi * 16];

    v8f acc = {};
#pragma unroll 2
    for (int k0 = 0; k0 < K; k0 += 32) {
        copy8_to_lds(aSrc + k0, aDst);
        copy8_to_lds(bSrc0 + k0, bDst0);
        copy8_to_lds(bSrc1 + k0, bDst1);
        tile_join();                       // s_wait_asynccnt + barrier
        v8h alo = *(const v8h*)aFrag;
        v8h ahi = *(const v8h*)(aFrag + 8);
        v16h a = __builtin_shufflevector(alo, ahi,
                 0, 1, 2, 3, 4, 5, 6, 7, 8, 9, 10, 11, 12, 13, 14, 15);
        v16h b = *(const v16h*)bFrag;
        acc = __builtin_amdgcn_wmma_f32_16x16x32_f16(
            false, a, false, b, (short)0, acc, false, false);
        __syncthreads();                   // reads done before next overwrite
    }

    const int col = tNblk + wx * 16 + l15;
    if (col < N) {
        float bv = bias ? bias[col] : 0.0f;
#pragma unroll
        for (int r = 0; r < 8; ++r) {
            int row = tMblk + wy * 16 + hi * 8 + r;
            out[(size_t)row * N + col] = acc[r] + bv;
        }
    }
}

// f32 -> f16 conversion (weights, once per use)
__global__ void k_f32_to_f16(const float* __restrict__ src,
                             _Float16* __restrict__ dst, int n) {
    int i = blockIdx.x * 256 + threadIdx.x;
    if (i < n) dst[i] = (_Float16)src[i];
}

// Encoder: out = gelu(LN(x @ w^T + b)) ; grid = B*T, block = 256 (= D)
__global__ void k_encode(const float* __restrict__ x, int F,
                         const float* __restrict__ w, const float* __restrict__ bvec,
                         const float* __restrict__ lg, const float* __restrict__ lb,
                         float* __restrict__ out) {
    __shared__ float xr[64];
    __shared__ float red[256];
    size_t row = blockIdx.x;
    int d = threadIdx.x;
    if (d < F) xr[d] = x[row * F + d];
    __syncthreads();
    float a = bvec[d];
    for (int f = 0; f < F; ++f) a += w[d * F + f] * xr[f];
    float mu  = blk_sum256(a, red) * (1.0f / 256.0f);
    float df  = a - mu;
    float var = blk_sum256(df * df, red) * (1.0f / 256.0f);
    float v   = df * rsqrtf(var + 1e-5f) * lg[d] + lb[d];
    out[row * 256 + d] = gelu_exact(v);
}

// LN -> f16 output (feeds in_proj GEMM only)
__global__ void k_ln_f16(const float* __restrict__ in,
                         const float* __restrict__ lg, const float* __restrict__ lb,
                         _Float16* __restrict__ out) {
    __shared__ float red[256];
    size_t row = blockIdx.x;
    int d = threadIdx.x;
    float a   = in[row * 256 + d];
    float mu  = blk_sum256(a, red) * (1.0f / 256.0f);
    float df  = a - mu;
    float var = blk_sum256(df * df, red) * (1.0f / 256.0f);
    out[row * 256 + d] = (_Float16)(df * rsqrtf(var + 1e-5f) * lg[d] + lb[d]);
}

// Depthwise causal conv (K=4) + SiLU; writes f32 (for scan) and f16 (for GEMM)
__global__ void k_conv_silu(const float* __restrict__ xz,
                            const float* __restrict__ cw, const float* __restrict__ cb,
                            float* __restrict__ xa, _Float16* __restrict__ xah, int T) {
    size_t row = blockIdx.x;
    int t = (int)(row % (size_t)T);
    for (int ci = 0; ci < 2; ++ci) {
        int c = ci * 256 + threadIdx.x;
        float a = cb[c];
#pragma unroll
        for (int k = 0; k < 4; ++k) {
            int tt = t - 3 + k;
            if (tt >= 0) a += cw[k * 512 + c] * xz[(row - 3 + k) * 1024 + c];
        }
        float v = siluf(a);
        xa[row * 512 + c]  = v;
        xah[row * 512 + c] = (_Float16)v;
    }
}

// delta = softplus(dt @ dt_w^T + dt_b) ; grid B*T, block 256
__global__ void k_dtproj(const float* __restrict__ dbc,
                         const float* __restrict__ dtw, const float* __restrict__ dtb,
                         float* __restrict__ delta) {
    size_t row = blockIdx.x;
    const float* dr = dbc + row * 48;
    for (int ci = 0; ci < 2; ++ci) {
        int d = ci * 256 + threadIdx.x;
        float a = dtb[d];
#pragma unroll
        for (int r = 0; r < 16; ++r) a += dr[r] * dtw[d * 16 + r];
        delta[row * 512 + d] = softplusf(a);
    }
}

// Selective-scan: one thread per (b,d), N=16 state in regs; grid=B, block=512
__global__ void k_scan(const float* __restrict__ delta,
                       const float* __restrict__ dbc,
                       const float* __restrict__ xa,
                       const float* __restrict__ A_log,
                       const float* __restrict__ Dp,
                       float* __restrict__ y, int T) {
    __shared__ float sBC[32];
    int b = blockIdx.x;
    int d = threadIdx.x;
    float A[16], h[16];
#pragma unroll
    for (int n = 0; n < 16; ++n) {
        A[n] = -__expf(A_log[d * 16 + n]);
        h[n] = 0.0f;
    }
    float Dd = Dp[d];
    for (int t = 0; t < T; ++t) {
        size_t row = (size_t)b * T + t;
        if (threadIdx.x < 32) sBC[threadIdx.x] = dbc[row * 48 + 16 + threadIdx.x];
        __syncthreads();
        float dt = delta[row * 512 + d];
        float xv = xa[row * 512 + d];
        float dx = dt * xv;
        float acc = 0.0f;
#pragma unroll
        for (int n = 0; n < 16; ++n) {
            h[n] = __expf(dt * A[n]) * h[n] + dx * sBC[n];
            acc += h[n] * sBC[16 + n];
        }
        y[row * 512 + d] = acc + xv * Dd;
        __syncthreads();
    }
}

// yh = f16( y * silu(z) ), z = xz[...,512:] ; feeds out_proj GEMM
__global__ void k_gate(const float* __restrict__ y, const float* __restrict__ xz,
                       _Float16* __restrict__ yh) {
    size_t i = (size_t)blockIdx.x * blockDim.x + threadIdx.x;
    size_t row = i >> 9;
    int c = (int)(i & 511);
    yh[i] = (_Float16)(y[i] * siluf(xz[row * 1024 + 512 + c]));
}

// x += res ; optional FiLM: x = g*x + b  (gb[row,0:256]=g, [256:512]=b)
__global__ void k_resfilm(float* __restrict__ x, const float* __restrict__ res,
                          const float* __restrict__ fgb) {
    size_t row = blockIdx.x;
    int d = threadIdx.x;
    float v = x[row * 256 + d] + res[row * 256 + d];
    if (fgb) v = fgb[row * 512 + d] * v + fgb[row * 512 + 256 + d];
    x[row * 256 + d] = v;
}

// Attention pooling over T ; grid = B, block = 256
__global__ void k_pool(const float* __restrict__ x,
                       const float* __restrict__ pw, const float* __restrict__ pb,
                       float* __restrict__ out, int T) {
    __shared__ float sc[1024];
    __shared__ float red[256];
    int b = blockIdx.x, tid = threadIdx.x;
    for (int t = tid; t < T; t += 256) {
        const float* row = x + ((size_t)b * T + t) * 256;
        float s = pb[0];
        for (int dd = 0; dd < 256; ++dd) s += row[dd] * pw[dd];
        sc[t] = s;
    }
    __syncthreads();
    float m = -1e30f;
    for (int t = tid; t < T; t += 256) m = fmaxf(m, sc[t]);
    red[tid] = m; __syncthreads();
    for (int o = 128; o > 0; o >>= 1) { if (tid < o) red[tid] = fmaxf(red[tid], red[tid + o]); __syncthreads(); }
    m = red[0]; __syncthreads();
    float ssum = 0.0f;
    for (int t = tid; t < T; t += 256) { float e = __expf(sc[t] - m); sc[t] = e; ssum += e; }
    float tot = blk_sum256(ssum, red);
    float inv = 1.0f / tot;
    float acc = 0.0f;
    for (int t = 0; t < T; ++t) acc += sc[t] * x[((size_t)b * T + t) * 256 + tid];
    out[b * 256 + tid] = acc * inv;
}

// FiLM macro MLP ; grid=B, block=256
__global__ void k_film_macro(const float* __restrict__ macro,
                             const float* __restrict__ mlng, const float* __restrict__ mlnb,
                             const float* __restrict__ w1, const float* __restrict__ b1,
                             const float* __restrict__ w2, const float* __restrict__ b2,
                             float* __restrict__ hmac) {
    __shared__ float m16[16];
    __shared__ float h1s[256];
    __shared__ float mv[2];
    int b = blockIdx.x, tid = threadIdx.x;
    if (tid < 16) m16[tid] = macro[b * 16 + tid];
    __syncthreads();
    if (tid == 0) {
        float m = 0.f; for (int i = 0; i < 16; ++i) m += m16[i]; m *= (1.f / 16.f);
        float v = 0.f; for (int i = 0; i < 16; ++i) { float df = m16[i] - m; v += df * df; }
        mv[0] = m; mv[1] = v * (1.f / 16.f);
    }
    __syncthreads();
    if (tid < 16) m16[tid] = (m16[tid] - mv[0]) * rsqrtf(mv[1] + 1e-5f) * mlng[tid] + mlnb[tid];
    __syncthreads();
    float a = b1[tid];
    for (int f = 0; f < 16; ++f) a += w1[tid * 16 + f] * m16[f];
    h1s[tid] = gelu_exact(a);
    __syncthreads();
    float a2 = b2[tid];
    for (int j = 0; j < 256; ++j) a2 += w2[tid * 256 + j] * h1s[j];
    hmac[b * 256 + tid] = a2;
}

// comb[b,t,d] = f16( hmac[b,d] + time-encoding ) ; grid = B*T, block = 256
__global__ void k_film_comb(const float* __restrict__ hmac,
                            const int* __restrict__ ts,
                            _Float16* __restrict__ comb, int T) {
    size_t row = blockIdx.x;
    int b = (int)(row / (size_t)T);
    int d = threadIdx.x;
    float tod = (float)(ts[row] % 86400);
    const float negc = -9.210340371976184f / 128.0f;  // -ln(1e4)/half
    float val;
    if (d < 128) { float fr = __expf((float)d * negc);         val = sinf(tod * fr); }
    else         { float fr = __expf((float)(d - 128) * negc); val = cosf(tod * fr); }
    comb[row * 256 + d] = (_Float16)(hmac[b * 256 + d] + val);
}

// Fusion attention + gate + 2x LN + head MLP ; grid = B, block = 256
__global__ void k_fusion_head(const float* __restrict__ sp_all,
                              const float* __restrict__ np_all,
                              const float* __restrict__ op_all,
                              const float* qw, const float* qb,
                              const float* kw, const float* kb,
                              const float* vw, const float* vb,
                              const float* ow, const float* ob,
                              const float* g1w, const float* g1b,
                              const float* g2w, const float* g2b,
                              const float* lng, const float* lnb,
                              const float* hlng, const float* hlnb,
                              const float* hw1, const float* hb1,
                              const float* hw2, const float* hb2,
                              float* __restrict__ outp) {
    __shared__ float sp[256], aux0[256], aux1[256];
    __shared__ float q[256], k0s[256], k1s[256], v0s[256], v1s[256];
    __shared__ float att[256], o[256], g1h[256], hln[256];
    __shared__ float red[256];
    __shared__ float s0[4], s1[4];
    __shared__ float h1[128];
    int b = blockIdx.x, tid = threadIdx.x;

    sp[tid]   = sp_all[b * 256 + tid];
    aux0[tid] = np_all[b * 256 + tid];
    aux1[tid] = op_all[b * 256 + tid];
    __syncthreads();

    float accq = qb[tid], ak0 = kb[tid], ak1 = kb[tid], av0 = vb[tid], av1 = vb[tid];
    for (int j = 0; j < 256; ++j) {
        float s  = sp[j], a0 = aux0[j], a1 = aux1[j];
        accq += qw[tid * 256 + j] * s;
        ak0  += kw[tid * 256 + j] * a0;
        ak1  += kw[tid * 256 + j] * a1;
        av0  += vw[tid * 256 + j] * a0;
        av1  += vw[tid * 256 + j] * a1;
    }
    q[tid] = accq; k0s[tid] = ak0; k1s[tid] = ak1; v0s[tid] = av0; v1s[tid] = av1;
    __syncthreads();

    red[tid] = q[tid] * k0s[tid]; __syncthreads();
    for (int off = 32; off > 0; off >>= 1) { if ((tid & 63) < off) red[tid] += red[tid + off]; __syncthreads(); }
    if ((tid & 63) == 0) s0[tid >> 6] = red[tid];
    __syncthreads();
    red[tid] = q[tid] * k1s[tid]; __syncthreads();
    for (int off = 32; off > 0; off >>= 1) { if ((tid & 63) < off) red[tid] += red[tid + off]; __syncthreads(); }
    if ((tid & 63) == 0) s1[tid >> 6] = red[tid];
    __syncthreads();

    int h = tid >> 6;
    float a0 = s0[h] * 0.125f, a1 = s1[h] * 0.125f;   // HD^-0.5
    float mx = fmaxf(a0, a1);
    float e0 = __expf(a0 - mx), e1 = __expf(a1 - mx);
    float inv = 1.0f / (e0 + e1);
    att[tid] = (e0 * v0s[tid] + e1 * v1s[tid]) * inv;
    __syncthreads();

    float ao = ob[tid];
    for (int j = 0; j < 256; ++j) ao += ow[tid * 256 + j] * att[j];
    o[tid] = ao;
    __syncthreads();

    float g1 = g1b[tid];
    for (int j = 0; j < 256; ++j) g1 += g1w[tid * 512 + j] * sp[j];
    for (int j = 0; j < 256; ++j) g1 += g1w[tid * 512 + 256 + j] * o[j];
    g1h[tid] = gelu_exact(g1);
    __syncthreads();

    float g2 = g2b[tid];
    for (int j = 0; j < 256; ++j) g2 += g2w[tid * 256 + j] * g1h[j];
    float gw = 1.0f / (1.0f + __expf(-g2));
    float fused = gw * o[tid] + (1.0f - gw) * sp[tid];

    float mu  = blk_sum256(fused, red) * (1.0f / 256.0f);
    float df  = fused - mu;
    float var = blk_sum256(df * df, red) * (1.0f / 256.0f);
    fused = df * rsqrtf(var + 1e-5f) * lng[tid] + lnb[tid];
    mu  = blk_sum256(fused, red) * (1.0f / 256.0f);
    df  = fused - mu;
    var = blk_sum256(df * df, red) * (1.0f / 256.0f);
    hln[tid] = df * rsqrtf(var + 1e-5f) * hlng[tid] + hlnb[tid];
    __syncthreads();

    if (tid < 128) {
        float a = hb1[tid];
        for (int j = 0; j < 256; ++j) a += hw1[tid * 256 + j] * hln[j];
        h1[tid] = gelu_exact(a);
    }
    __syncthreads();
    red[tid] = (tid < 128) ? h1[tid] * hw2[tid] : 0.0f;
    __syncthreads();
    for (int off = 128; off > 0; off >>= 1) { if (tid < off) red[tid] += red[tid + off]; __syncthreads(); }
    if (tid == 0) outp[b] = red[0] + hb2[0];
}

// ---------------------------------------------------------------------------
// Host orchestration
// ---------------------------------------------------------------------------
extern "C" void kernel_launch(void* const* d_in, const int* in_sizes, int n_in,
                              void* d_out, int out_size, void* d_ws, size_t ws_size,
                              hipStream_t stream) {
    (void)in_sizes; (void)n_in; (void)out_size; (void)ws_size;

    auto P = [&](int i) -> const float* { return (const float*)d_in[i]; };
    const int* ts = (const int*)d_in[4];

    // params flattening: d_in[0]=stock_x,[1]=options_x,[2]=news_x,[3]=macro,
    // [4]=timestamps; per stream (stock,news,options) 28 entries; film@89,
    // fg@99, head@113.
    const int FB = 5 + 3 * 28;   // 89
    const int GB = FB + 10;      // 99
    const int HB = GB + 14;      // 113

    const int M = 4096;          // B*T
    float* ws = (float*)d_ws;
    const size_t S_H  = (size_t)M * 256;
    const size_t S_XZ = (size_t)M * 1024;
    const size_t S_XA = (size_t)M * 512;
    const size_t S_DBC = (size_t)M * 48;
    float* hb[3]   = { ws, ws + S_H, ws + 2 * S_H };
    float* t_res   = ws + 3 * S_H;        // out-proj result (f32)
    float* t_xz    = t_res + S_H;
    float* t_xa    = t_xz + S_XZ;
    float* t_dbc   = t_xa + S_XA;
    float* t_delta = t_dbc + S_DBC;
    float* t_y     = t_delta + S_XA;
    float* t_fgb0  = t_y + S_XA;
    float* t_fgb1  = t_fgb0 + S_XA;
    float* t_hmac  = t_fgb1 + S_XA;
    float* t_pool  = t_hmac + 1024;
    // f16 region (2-byte elems), 16B-aligned by construction
    _Float16* t_lnh   = (_Float16*)(t_pool + 4096);
    _Float16* t_xah   = t_lnh + S_H;
    _Float16* t_yh    = t_xah + S_XA;
    _Float16* t_combh = t_yh + S_XA;
    _Float16* t_wh    = t_combh + S_H;    // max 1024*256 halves

    auto cvt = [&](const float* src, _Float16* dst, int n) {
        k_f32_to_f16<<<(n + 255) / 256, 256, 0, stream>>>(src, dst, n);
    };
    auto gemm256 = [&](const _Float16* A, const _Float16* W, const float* bias,
                       float* out, int m, int n) {
        dim3 g(m / 32, (n + 63) / 64);
        k_gemm_wmma<256><<<g, 256, 0, stream>>>(A, W, bias, out, m, n);
    };
    auto gemm512 = [&](const _Float16* A, const _Float16* W, const float* bias,
                       float* out, int m, int n) {
        dim3 g(m / 32, (n + 63) / 64);
        k_gemm_wmma<512><<<g, 256, 0, stream>>>(A, W, bias, out, m, n);
    };

    // ---- FiLM ----
    k_film_macro<<<4, 256, 0, stream>>>(P(3), P(FB + 0), P(FB + 1), P(FB + 2),
                                        P(FB + 3), P(FB + 4), P(FB + 5), t_hmac);
    k_film_comb<<<M, 256, 0, stream>>>(t_hmac, ts, t_combh, 1024);
    cvt(P(FB + 6), t_wh, 512 * 256);
    gemm256(t_combh, t_wh, P(FB + 7), t_fgb0, M, 512);
    cvt(P(FB + 8), t_wh, 512 * 256);
    gemm256(t_combh, t_wh, P(FB + 9), t_fgb1, M, 512);

    // ---- streams: param order (stock,news,options); x order (stock,options,news)
    const float* xin[3] = { P(0), P(2), P(1) };
    const int    Fdim[3] = { 20, 64, 38 };

    for (int s = 0; s < 3; ++s) {
        int pb = 5 + s * 28;
        k_encode<<<M, 256, 0, stream>>>(xin[s], Fdim[s], P(pb + 0), P(pb + 1),
                                        P(pb + 2), P(pb + 3), hb[s]);
        for (int l = 0; l < 2; ++l) {
            int lb = pb + 4 + l * 11;
            k_ln_f16<<<M, 256, 0, stream>>>(hb[s], P(lb + 9), P(lb + 10), t_lnh);
            cvt(P(lb + 0), t_wh, 1024 * 256);
            gemm256(t_lnh, t_wh, nullptr, t_xz, M, 1024);                 // in_proj
            k_conv_silu<<<M, 256, 0, stream>>>(t_xz, P(lb + 1), P(lb + 2),
                                               t_xa, t_xah, 1024);
            cvt(P(lb + 3), t_wh, 48 * 512);
            gemm512(t_xah, t_wh, nullptr, t_dbc, M, 48);                  // xproj
            k_dtproj<<<M, 256, 0, stream>>>(t_dbc, P(lb + 4), P(lb + 5), t_delta);
            k_scan<<<4, 512, 0, stream>>>(t_delta, t_dbc, t_xa, P(lb + 6), P(lb + 7),
                                          t_y, 1024);
            k_gate<<<(M * 512) / 256, 256, 0, stream>>>(t_y, t_xz, t_yh);
            cvt(P(lb + 8), t_wh, 256 * 512);
            gemm512(t_yh, t_wh, nullptr, t_res, M, 256);                  // out_proj
            const float* fgb = (s == 0) ? ((l == 0) ? t_fgb0 : t_fgb1) : nullptr;
            k_resfilm<<<M, 256, 0, stream>>>(hb[s], t_res, fgb);
        }
        k_pool<<<4, 256, 0, stream>>>(hb[s], P(pb + 26), P(pb + 27),
                                      t_pool + s * 1024, 1024);
    }

    // ---- fusion + head ----
    k_fusion_head<<<4, 256, 0, stream>>>(
        t_pool, t_pool + 1024, t_pool + 2048,
        P(GB + 0), P(GB + 1), P(GB + 2), P(GB + 3),
        P(GB + 4), P(GB + 5), P(GB + 6), P(GB + 7),
        P(GB + 8), P(GB + 9), P(GB + 10), P(GB + 11),
        P(GB + 12), P(GB + 13),
        P(HB + 0), P(HB + 1), P(HB + 2), P(HB + 3), P(HB + 4), P(HB + 5),
        (float*)d_out);
}
